// ProjectiveMeasurement_24043226923419
// MI455X (gfx1250) — matrix operand order repaired and checked by reading, not compiled, not run on hardware
//
#include <hip/hip_runtime.h>
#include <stdint.h>

#define DIM      4096
#define NUM_OUT  2048
#define BATCH    8192

typedef __attribute__((ext_vector_type(16))) __bf16 v16bf;
typedef __attribute__((ext_vector_type(8)))  float  v8f;

// ---------------- JAX threefry2x32 (20 rounds), key = jax.random.key(42) = (0,42)
__device__ __forceinline__ uint32_t rotl32(uint32_t x, int r) { return (x << r) | (x >> (32 - r)); }

__device__ __forceinline__ void threefry2x32(uint32_t k0, uint32_t k1,
                                             uint32_t x0, uint32_t x1,
                                             uint32_t& o0, uint32_t& o1) {
  uint32_t ks0 = k0, ks1 = k1, ks2 = k0 ^ k1 ^ 0x1BD11BDAu;
  x0 += ks0; x1 += ks1;
#define TF_R(r) { x0 += x1; x1 = rotl32(x1, r); x1 ^= x0; }
  TF_R(13) TF_R(15) TF_R(26) TF_R(6)
  x0 += ks1; x1 += ks2 + 1u;
  TF_R(17) TF_R(29) TF_R(16) TF_R(24)
  x0 += ks2; x1 += ks0 + 2u;
  TF_R(13) TF_R(15) TF_R(26) TF_R(6)
  x0 += ks0; x1 += ks1 + 3u;
  TF_R(17) TF_R(29) TF_R(16) TF_R(24)
  x0 += ks1; x1 += ks2 + 4u;
  TF_R(13) TF_R(15) TF_R(26) TF_R(6)
  x0 += ks2; x1 += ks0 + 5u;
#undef TF_R
  o0 = x0; o1 = x1;
}

// JAX gumbel sample at flat index j of a (BATCH*NUM_OUT) draw with key (0,42):
// counts are iota split in halves -> pair (j, j+h) first word / (j-h, j) second word.
__device__ __forceinline__ float gumbel_at(uint64_t j) {
  const uint64_t n = (uint64_t)BATCH * (uint64_t)NUM_OUT;
  const uint64_t h = n >> 1;
  uint32_t o0, o1, bits;
  if (j < h) { threefry2x32(0u, 42u, (uint32_t)j,        (uint32_t)(j + h), o0, o1); bits = o0; }
  else       { threefry2x32(0u, 42u, (uint32_t)(j - h),  (uint32_t)j,       o0, o1); bits = o1; }
  float f = __uint_as_float((bits >> 9) | 0x3f800000u) - 1.0f;   // [0,1)
  float u = fmaxf(1.1754944e-38f, f);                             // minval = tiny
  return -__logf(-__logf(u));
}

// order-preserving float -> u32 map (works for negatives)
__device__ __forceinline__ uint32_t f2ord(float f) {
  uint32_t s = __float_as_uint(f);
  return (s & 0x80000000u) ? ~s : (s | 0x80000000u);
}

// ---------------- workspace init (must be deterministic every launch)
__global__ void pm_init_ws(unsigned long long* __restrict__ ws) {
  int i = blockIdx.x * 256 + threadIdx.x;
  if (i < BATCH) ws[i] = 0ull;
}

// ---------------- main GEMM + sampling kernel
// grid: x = NUM_OUT/128, y = BATCH/64 ; 256 threads = 8 waves (wave32)
// wave (waveB in 0..1, waveO in 0..3) computes a 32(b) x 32(o) tile = 2x2 WMMA tiles
__global__ __launch_bounds__(256)
void pm_probs_sample(const float* __restrict__ state,    // [B, DIM, 2]
                     const float* __restrict__ basis,    // [DIM, O, 2]
                     unsigned long long* __restrict__ rowbest) {
  __shared__ unsigned long long lmax[64];

  const int tid   = threadIdx.x;
  const int lane  = tid & 31;
  const int wave  = tid >> 5;
  const int waveB = wave >> 2;   // 0..1
  const int waveO = wave & 3;    // 0..3
  const int blockB = blockIdx.y * 64;
  const int blockO = blockIdx.x * 128;

  if (tid < 64) lmax[tid] = 0ull;
  __syncthreads();

  const int m = lane & 15;   // row (A) / col (B) within 16
  const int h = lane >> 4;   // half-wave select

  v8f accR[2][2], accI[2][2], accP[2][2];
#pragma unroll
  for (int i = 0; i < 2; ++i)
#pragma unroll
    for (int j = 0; j < 2; ++j) { accR[i][j] = (v8f)0.0f; accI[i][j] = (v8f)0.0f; accP[i][j] = (v8f)0.0f; }

  const float2* st = (const float2*)state;   // [B][DIM] complex
  const float2* bs = (const float2*)basis;   // [DIM][O] complex

  const int b0 = blockB + waveB * 32;
  const int o0 = blockO + waveO * 32;

  for (int kk = 0; kk < DIM; kk += 32) {
    // ---- A fragments: 16x32 bf16; lane(m,h): VGPR0..3 -> K = 8h + 0..7, VGPR4..7 -> K = 16 + 8h + 0..7
    v16bf ar[2], ai[2];
#pragma unroll
    for (int sb = 0; sb < 2; ++sb) {
      const float2* srow = st + (size_t)(b0 + sb * 16 + m) * DIM;
      const int dlo = kk + 8 * h;
      const int dhi = kk + 16 + 8 * h;
#pragma unroll
      for (int j = 0; j < 8; ++j) {
        float2 lo = srow[dlo + j];
        float2 hi = srow[dhi + j];
        ar[sb][j]     = (__bf16)lo.x;  ai[sb][j]     = (__bf16)lo.y;
        ar[sb][8 + j] = (__bf16)hi.x;  ai[sb][8 + j] = (__bf16)hi.y;
      }
    }
    // ---- B fragments: 32x16 bf16; lane(n=m,h): VGPR0..3 -> K = 8h + 0..7, VGPR4..7 -> K = 16 + 8h + 0..7
    v16bf br[2], bi[2];
#pragma unroll
    for (int so = 0; so < 2; ++so) {
      const float2* brow = bs + (size_t)(kk + 8 * h) * NUM_OUT + (o0 + so * 16 + m);
#pragma unroll
      for (int t = 0; t < 16; ++t) {
        float2 v = brow[(size_t)((t & 7) + 16 * (t >> 3)) * NUM_OUT];
        br[so][t] = (__bf16)v.x;  bi[so][t] = (__bf16)v.y;
      }
    }
    // ---- 16 WMMAs per K=32 step
#pragma unroll
    for (int sb = 0; sb < 2; ++sb)
#pragma unroll
      for (int so = 0; so < 2; ++so) {
        // accR += si*bi ; accR += sr*br   (inner_real)
        accR[sb][so] = __builtin_amdgcn_wmma_f32_16x16x32_bf16(false, ai[sb], false, bi[so],
                                                               (short)0, accR[sb][so], false, false);
        accR[sb][so] = __builtin_amdgcn_wmma_f32_16x16x32_bf16(false, ar[sb], false, br[so],
                                                               (short)0, accR[sb][so], false, false);
        // accI += si*br ; accP += sr*bi   (inner_imag = accI - accP)
        accI[sb][so] = __builtin_amdgcn_wmma_f32_16x16x32_bf16(false, ai[sb], false, br[so],
                                                               (short)0, accI[sb][so], false, false);
        accP[sb][so] = __builtin_amdgcn_wmma_f32_16x16x32_bf16(false, ar[sb], false, bi[so],
                                                               (short)0, accP[sb][so], false, false);
      }
  }

  // ---- epilogue: prob -> logit -> +gumbel -> packed per-row argmax (LDS then global)
#pragma unroll
  for (int sb = 0; sb < 2; ++sb)
#pragma unroll
    for (int so = 0; so < 2; ++so)
#pragma unroll
      for (int j = 0; j < 8; ++j) {
        float re = accR[sb][so][j];
        float im = accI[sb][so][j] - accP[sb][so][j];
        float prob  = re * re + im * im;
        float logit = __logf(prob + 1e-10f);
        const int rloc = waveB * 32 + sb * 16 + j + 8 * h;   // 0..63 within block
        const int b = blockB + rloc;
        const int o = o0 + so * 16 + m;
        float key = logit + gumbel_at((uint64_t)b * NUM_OUT + (uint64_t)o);
        unsigned long long packed =
            ((unsigned long long)f2ord(key) << 32) | (unsigned long long)(uint32_t)o;
        atomicMax(&lmax[rloc], packed);
      }
  __syncthreads();
  if (tid < 64) atomicMax(&rowbest[blockB + tid], lmax[tid]);
}

// ---------------- outcome write + collapsed gather
__global__ __launch_bounds__(256)
void pm_collapse(const float* __restrict__ basis,
                 const unsigned long long* __restrict__ rowbest,
                 float* __restrict__ out) {
  __shared__ int s_o;
  const int b = blockIdx.x;
  if (threadIdx.x == 0) {
    unsigned long long p = rowbest[b];
    int o = (int)(uint32_t)(p & 0xFFFFFFFFull);
    s_o = o;
    out[b] = (float)o;                 // outcome[b]
  }
  __syncthreads();
  const int o = s_o;
  const float2* bs = (const float2*)basis;           // [DIM][O] complex
  float2* co = (float2*)(out + BATCH);               // collapsed [B][DIM] complex
  for (int d = threadIdx.x; d < DIM; d += 256)
    co[(size_t)b * DIM + d] = bs[(size_t)d * NUM_OUT + o];
}

extern "C" void kernel_launch(void* const* d_in, const int* in_sizes, int n_in,
                              void* d_out, int out_size, void* d_ws, size_t ws_size,
                              hipStream_t stream) {
  const float* state = (const float*)d_in[0];   // [BATCH, DIM, 2] f32
  const float* basis = (const float*)d_in[1];   // [DIM, NUM_OUT, 2] f32
  float* out = (float*)d_out;                   // [BATCH] outcome + [BATCH, DIM, 2] collapsed
  unsigned long long* ws = (unsigned long long*)d_ws;  // BATCH packed argmax slots (64 KB)

  pm_init_ws<<<(BATCH + 255) / 256, 256, 0, stream>>>(ws);

  dim3 grid(NUM_OUT / 128, BATCH / 64);
  pm_probs_sample<<<grid, 256, 0, stream>>>(state, basis, ws);

  pm_collapse<<<BATCH, 256, 0, stream>>>(basis, ws, out);
}
